// RelationalTransformerUpdate_45466523795987
// MI455X (gfx1250) — matrix operand order repaired and checked by reading, not compiled, not run on hardware
//
#include <hip/hip_runtime.h>
#include <hip/hip_bf16.h>
#include <cstring>
#include <cstdint>

// ---------------------------------------------------------------------------
// Problem constants (from the reference)
// ---------------------------------------------------------------------------
namespace rt {
constexpr int B = 2, S = 512, Dm = 1024, H = 8, L = 2, FF = 4096, NR = 37, DK = 128;
constexpr float SCALE = 0.08838834764831845f;  // 1/sqrt(DK)
}

typedef __attribute__((ext_vector_type(16))) __bf16 bf16x16;
typedef __attribute__((ext_vector_type(8)))  float  f32x8;

__device__ __forceinline__ unsigned short f2bf(float f) {
  union { float f; unsigned u; } a; a.f = f;
  unsigned u = a.u;
  unsigned r = u + 0x7FFFu + ((u >> 16) & 1u);  // round-to-nearest-even
  return (unsigned short)(r >> 16);
}

// CDNA5 async global->LDS copies (16B/lane/beat, ASYNCcnt-tracked).
// IOFFSET is applied to BOTH the LDS dest and global src addresses (ISA 10.7),
// so multi-beat copies share one base address pair.
__device__ __forceinline__ void async_ld_b128_x4(unsigned lds_byte_addr, const void* gaddr) {
  asm volatile(
      "global_load_async_to_lds_b128 %0, %1, off\n\t"
      "global_load_async_to_lds_b128 %0, %1, off offset:16\n\t"
      "global_load_async_to_lds_b128 %0, %1, off offset:32\n\t"
      "global_load_async_to_lds_b128 %0, %1, off offset:48"
      :: "v"(lds_byte_addr), "v"(gaddr) : "memory");
}
__device__ __forceinline__ void async_ld_b128_x2(unsigned lds_byte_addr, const void* gaddr) {
  asm volatile(
      "global_load_async_to_lds_b128 %0, %1, off\n\t"
      "global_load_async_to_lds_b128 %0, %1, off offset:16"
      :: "v"(lds_byte_addr), "v"(gaddr) : "memory");
}
__device__ __forceinline__ void wait_async0() {
  asm volatile("s_wait_asynccnt 0x0" ::: "memory");
}
__device__ __forceinline__ unsigned lds_addr_of(const void* p) {
  return (unsigned)(unsigned long long)p;  // LDS aperture: low 32 bits = LDS byte offset
}

// ---------------------------------------------------------------------------
// Block reductions (wave32)
// ---------------------------------------------------------------------------
__device__ __forceinline__ float block_sum(float v, float* sbuf) {
  __syncthreads();
  #pragma unroll
  for (int o = 16; o > 0; o >>= 1) v += __shfl_down(v, o, 32);
  int wave = threadIdx.x >> 5, lane = threadIdx.x & 31;
  if (lane == 0) sbuf[wave] = v;
  __syncthreads();
  if (wave == 0) {
    float r = (lane < (int)(blockDim.x >> 5)) ? sbuf[lane] : 0.f;
    #pragma unroll
    for (int o = 4; o > 0; o >>= 1) r += __shfl_down(r, o, 32);
    if (lane == 0) sbuf[0] = r;
  }
  __syncthreads();
  return sbuf[0];
}

__device__ __forceinline__ float block_max(float v, float* sbuf) {
  __syncthreads();
  #pragma unroll
  for (int o = 16; o > 0; o >>= 1) v = fmaxf(v, __shfl_down(v, o, 32));
  int wave = threadIdx.x >> 5, lane = threadIdx.x & 31;
  if (lane == 0) sbuf[wave] = v;
  __syncthreads();
  if (wave == 0) {
    float r = (lane < (int)(blockDim.x >> 5)) ? sbuf[lane] : -3.4e38f;
    #pragma unroll
    for (int o = 4; o > 0; o >>= 1) r = fmaxf(r, __shfl_down(r, o, 32));
    if (lane == 0) sbuf[0] = r;
  }
  __syncthreads();
  return sbuf[0];
}

// ---------------------------------------------------------------------------
// fp32 -> bf16 conversion (grid-stride)
// ---------------------------------------------------------------------------
__global__ __launch_bounds__(256) void f32_to_bf16_kernel(const float* in, unsigned short* out, long long n) {
  long long i = (long long)blockIdx.x * blockDim.x + threadIdx.x;
  long long stride = (long long)gridDim.x * blockDim.x;
  for (; i < n; i += stride) out[i] = f2bf(in[i]);
}

// ---------------------------------------------------------------------------
// [K,N] fp32 -> [N,K] bf16 transposed conversion (32x32 LDS tiles, z = layer)
// ---------------------------------------------------------------------------
__global__ __launch_bounds__(256) void transpose_cvt_kernel(const float* in, unsigned short* out,
                                                            int K, int N) {
  __shared__ unsigned short tile[32][33];
  const long long lbase = (long long)blockIdx.z * K * N;
  int tx = threadIdx.x & 31;
  int ty = threadIdx.x >> 5;          // 0..7
  int k0 = blockIdx.y * 32;
  int n0 = blockIdx.x * 32;
  #pragma unroll
  for (int r = 0; r < 4; ++r) {
    int kk = k0 + ty + r * 8;
    tile[ty + r * 8][tx] = f2bf(in[lbase + (long long)kk * N + (n0 + tx)]);
  }
  __syncthreads();
  #pragma unroll
  for (int r = 0; r < 4; ++r) {
    int nn = n0 + ty + r * 8;
    out[lbase + (long long)nn * K + (k0 + tx)] = tile[tx][ty + r * 8];
  }
}

// ---------------------------------------------------------------------------
// LayerNorm (annotated-transformer style: unbiased std, eps added to std)
// ---------------------------------------------------------------------------
__global__ __launch_bounds__(256) void ln_kernel(const float* x, const float* g, const float* bta,
                                                 unsigned short* obf, float* of, int Dmodel) {
  __shared__ float sbuf[8];
  long long row = blockIdx.x;
  const float* xr = x + row * Dmodel;
  float s = 0.f, s2 = 0.f;
  for (int i = threadIdx.x; i < Dmodel; i += blockDim.x) { float v = xr[i]; s += v; s2 += v * v; }
  float S1 = block_sum(s, sbuf);
  float S2 = block_sum(s2, sbuf);
  float mean = S1 / Dmodel;
  float var = (S2 - S1 * mean) / (Dmodel - 1);
  var = fmaxf(var, 0.f);
  float inv = 1.f / (sqrtf(var) + 1e-6f);
  for (int i = threadIdx.x; i < Dmodel; i += blockDim.x) {
    float v = g[i] * (xr[i] - mean) * inv + bta[i];
    if (obf) obf[row * Dmodel + i] = f2bf(v);
    if (of)  of[row * Dmodel + i] = v;
  }
}

// ---------------------------------------------------------------------------
// Batched bf16 WMMA GEMM, async-LDS double-buffered, B always given as [N,K].
//   C(fp32) = A[M,K] x B^T (+bias); accum: C += ; relu fused
//   Cb: optional bf16 row-major mirror; Ct: optional bf16 transposed [N,M] out
// ---------------------------------------------------------------------------
struct GemmP {
  const unsigned short* A; int lda; long long A_bs, A_hs;
  const unsigned short* Bm; int ldb; long long B_bs, B_hs;   // [N][K]
  float* C; int ldc; long long C_bs, C_hs;
  unsigned short* Cb; int ldcb; long long Cb_bs, Cb_hs;
  unsigned short* Ct; int ldt; long long Ct_bs, Ct_hs;       // [N][M]
  const float* bias;
  int M, N, K, Hdiv, relu, accum;
};

#define GBM 128
#define GBN 64
#define GBK 64
#define LDSA 72   // padded LDS row stride (halfs)
#define LDSB 72

__global__ __launch_bounds__(256) void gemm_bf16_kernel(GemmP p) {
  __shared__ unsigned short sA[2][GBM * LDSA];
  __shared__ unsigned short sB[2][GBN * LDSB];
  const int tid = threadIdx.x;
  const int bz = blockIdx.z;
  const int bh = bz % p.Hdiv;
  const int bb = bz / p.Hdiv;
  const unsigned short* A  = p.A  + (long long)bb * p.A_bs + (long long)bh * p.A_hs;
  const unsigned short* Bm = p.Bm + (long long)bb * p.B_bs + (long long)bh * p.B_hs;
  const long long coff  = (long long)bb * p.C_bs  + (long long)bh * p.C_hs;
  const long long cboff = (long long)bb * p.Cb_bs + (long long)bh * p.Cb_hs;
  const long long ctoff = (long long)bb * p.Ct_bs + (long long)bh * p.Ct_hs;
  const int m0 = blockIdx.y * GBM;
  const int n0 = blockIdx.x * GBN;
  const int lane = tid & 31, wave = tid >> 5;
  const int wm = wave & 3, wn = wave >> 2;      // 4x2 wave grid -> 128x64
  const int half = lane >> 4, l16 = lane & 15;
  const bool fullM = (m0 + GBM <= p.M);   // uniform per block
  const bool fullN = (n0 + GBN <= p.N);   // uniform per block

  f32x8 acc[2][2] = {};

  // stage a (GBM x GBK) A tile and (GBN x GBK) B tile into buffer `buf`
  auto load_tiles = [&](int k0, int buf) {
    {  // ---- A tile: 128 rows x 64 halfs; thread: 64B of one row
      int r = tid >> 1, cb = (tid & 1) * 32;
      const unsigned short* src = A + (long long)(m0 + r) * p.lda + (k0 + cb);
      unsigned la = lds_addr_of(&sA[buf][r * LDSA + cb]);
      if (fullM) {
        async_ld_b128_x4(la, src);
      } else {
        uint4 d[4] = {make_uint4(0,0,0,0), make_uint4(0,0,0,0),
                      make_uint4(0,0,0,0), make_uint4(0,0,0,0)};
        if (m0 + r < p.M) {
          #pragma unroll
          for (int c = 0; c < 4; ++c) d[c] = *(const uint4*)(src + c * 8);
        }
        #pragma unroll
        for (int c = 0; c < 4; ++c) *(uint4*)&sA[buf][r * LDSA + cb + c * 8] = d[c];
      }
    }
    {  // ---- B tile: 64 rows x 64 halfs; thread: 32B of one row
      int n = tid >> 2, cb = (tid & 3) * 16;
      const unsigned short* src = Bm + (long long)(n0 + n) * p.ldb + (k0 + cb);
      unsigned lb = lds_addr_of(&sB[buf][n * LDSB + cb]);
      if (fullN) {
        async_ld_b128_x2(lb, src);
      } else {
        uint4 d0 = make_uint4(0,0,0,0), d1 = make_uint4(0,0,0,0);
        if (n0 + n < p.N) { d0 = *(const uint4*)src; d1 = *(const uint4*)(src + 8); }
        *(uint4*)&sB[buf][n * LDSB + cb]     = d0;
        *(uint4*)&sB[buf][n * LDSB + cb + 8] = d1;
      }
    }
  };

  const int nk = p.K / GBK;
  load_tiles(0, 0);
  wait_async0();
  __syncthreads();

  for (int it = 0; it < nk; ++it) {
    const int cur = it & 1;
    if (it + 1 < nk) load_tiles((it + 1) * GBK, cur ^ 1);  // async prefetch next stage

    union V16 { uint4 q[2]; bf16x16 v; };
    #pragma unroll
    for (int ks = 0; ks < 2; ++ks) {
      V16 afr[2], bfr[2];
      #pragma unroll
      for (int fi = 0; fi < 2; ++fi) {
        // A 16x32: lane m = l16, K chunks [half*8,+8) and [16+half*8,+8)
        const unsigned short* ap =
            &sA[cur][(wm * 32 + fi * 16 + l16) * LDSA + ks * 32 + half * 8];
        afr[fi].q[0] = *(const uint4*)ap;
        afr[fi].q[1] = *(const uint4*)(ap + 16);
      }
      #pragma unroll
      for (int fj = 0; fj < 2; ++fj) {
        // B 32x16: lane n = l16, K chunks [half*8,+8) and [16+half*8,+8)
        const unsigned short* bp =
            &sB[cur][(wn * 32 + fj * 16 + l16) * LDSB + ks * 32 + half * 8];
        bfr[fj].q[0] = *(const uint4*)bp;
        bfr[fj].q[1] = *(const uint4*)(bp + 16);
      }
      #pragma unroll
      for (int fi = 0; fi < 2; ++fi)
        #pragma unroll
        for (int fj = 0; fj < 2; ++fj)
          acc[fi][fj] = __builtin_amdgcn_wmma_f32_16x16x32_bf16(
              false, afr[fi].v, false, bfr[fj].v, (short)0, acc[fi][fj], false, false);
    }

    wait_async0();      // this wave's prefetch into the other buffer landed
    __syncthreads();    // everyone done reading `cur` / writing `cur^1`
  }

  // ---- epilogue: C/D layout: lane n = l16; vgpr r -> m = r + half*8
  #pragma unroll
  for (int fi = 0; fi < 2; ++fi) {
    #pragma unroll
    for (int fj = 0; fj < 2; ++fj) {
      int n = n0 + wn * 32 + fj * 16 + l16;
      if (n >= p.N) continue;
      float bsv = p.bias ? p.bias[n] : 0.f;
      int mb = m0 + wm * 32 + fi * 16 + half * 8;
      unsigned short tpack[8];
      #pragma unroll
      for (int r = 0; r < 8; ++r) {
        int m = mb + r;
        if (m >= p.M) continue;
        float v = acc[fi][fj][r] + bsv;
        long long idx = coff + (long long)m * p.ldc + n;
        if (p.accum) v += p.C[idx];
        if (p.relu && v < 0.f) v = 0.f;
        if (p.C)  p.C[idx] = v;
        if (p.Cb) p.Cb[cboff + (long long)m * p.ldcb + n] = f2bf(v);
        tpack[r] = f2bf(v);
      }
      if (p.Ct && mb + 8 <= p.M)
        *(uint4*)&p.Ct[ctoff + (long long)n * p.ldt + mb] = *(const uint4*)tpack;
    }
  }
}

// ---------------------------------------------------------------------------
// Fused: logits' = (qk + qr[rel]) * scale ; softmax ; attn -> bf16
// plus per-relation attention mass aw[row][r] = sum_j attn[j] * (rel[j]==r)
// ---------------------------------------------------------------------------
__global__ __launch_bounds__(256) void softmax_rel_kernel(const float* logits, const float* qr,
                                                          const int* rel, unsigned short* attnb,
                                                          float* aw, float scale) {
  using namespace rt;
  __shared__ float sqr[NR];
  __shared__ float saw[NR];
  __shared__ float sbuf[8];
  int row = blockIdx.x;                 // (b*H + h)*S + i
  int i  = row & (S - 1);
  int bh = row / S;
  int b  = bh / H;
  const float* lg  = logits + (long long)row * S;
  const int*   rr  = rel + ((long long)b * S + i) * S;
  const float* qrr = qr + (long long)row * NR;
  int tid = threadIdx.x;
  if (tid < NR) { sqr[tid] = qrr[tid]; saw[tid] = 0.f; }
  __syncthreads();
  int j0 = tid, j1 = tid + 256;
  int r0 = rr[j0], r1 = rr[j1];
  float t0 = (lg[j0] + sqr[r0]) * scale;
  float t1 = (lg[j1] + sqr[r1]) * scale;
  float mx = block_max(fmaxf(t0, t1), sbuf);
  float e0 = __expf(t0 - mx), e1 = __expf(t1 - mx);
  float se = block_sum(e0 + e1, sbuf);
  float inv = 1.f / se;
  float a0 = e0 * inv, a1 = e1 * inv;
  attnb[(long long)row * S + j0] = f2bf(a0);
  attnb[(long long)row * S + j1] = f2bf(a1);
  atomicAdd(&saw[r0], a0);
  atomicAdd(&saw[r1], a1);
  __syncthreads();
  if (tid < NR) aw[(long long)row * NR + tid] = saw[tid];
}

// ---------------------------------------------------------------------------
// ctx_out[b,i,h*DK+d] = ctx + sum_r aw[b,h,i,r] * rel_v[r,d]  -> bf16
// ---------------------------------------------------------------------------
__global__ __launch_bounds__(128) void add_rv_kernel(const float* ctx, unsigned short* outb,
                                                     const float* aw, const float* relv) {
  using namespace rt;
  __shared__ float srel[NR * DK];
  __shared__ float sa[8][NR];
  int tid = threadIdx.x;
  for (int idx = tid; idx < NR * DK; idx += 128) srel[idx] = relv[idx];
  long long row0 = (long long)blockIdx.x * 8;
  for (int rr2 = tid; rr2 < 8 * NR; rr2 += 128) {
    int r8 = rr2 / NR, r = rr2 % NR;
    sa[r8][r] = aw[(row0 + r8) * NR + r];
  }
  __syncthreads();
  int bh = (int)(row0 / S);
  int i0 = (int)(row0 % S);
  int h = bh % H, b = bh / H;
  for (int r8 = 0; r8 < 8; ++r8) {
    float s = 0.f;
    #pragma unroll
    for (int r = 0; r < NR; ++r) s += sa[r8][r] * srel[r * DK + tid];
    long long o = ((long long)b * S + (i0 + r8)) * Dm + (long long)h * DK + tid;
    outb[o] = f2bf(ctx[o] + s);
  }
}

// ---------------------------------------------------------------------------
// Host orchestration
// ---------------------------------------------------------------------------
extern "C" void kernel_launch(void* const* d_in, const int* in_sizes, int n_in,
                              void* d_out, int out_size, void* d_ws, size_t ws_size,
                              hipStream_t stream) {
  (void)in_sizes; (void)n_in; (void)out_size; (void)ws_size;
  using namespace rt;
  const float* x    = (const float*)d_in[0];
  const int*   rel  = (const int*)d_in[1];
  const float* bq   = (const float*)d_in[3];
  const float* bk   = (const float*)d_in[5];
  const float* bv   = (const float*)d_in[7];
  const float* bo   = (const float*)d_in[9];
  const float* ln1g = (const float*)d_in[10]; const float* ln1b = (const float*)d_in[11];
  const float* ln2g = (const float*)d_in[12]; const float* ln2b = (const float*)d_in[13];
  const float* b1   = (const float*)d_in[15];
  const float* b2   = (const float*)d_in[17];
  const float* rel_v = (const float*)d_in[19];
  const float* lnfg = (const float*)d_in[20]; const float* lnfb = (const float*)d_in[21];

  char* w = (char*)d_ws;
  auto carve = [&](size_t bytes) -> char* {
    char* p = w; w += (bytes + 255) & ~(size_t)255; return p;
  };
  const long long MD = (long long)B * S * Dm;
  unsigned short* WQT = (unsigned short*)carve((size_t)L * Dm * Dm * 2);
  unsigned short* WKT = (unsigned short*)carve((size_t)L * Dm * Dm * 2);
  unsigned short* WVT = (unsigned short*)carve((size_t)L * Dm * Dm * 2);
  unsigned short* WOT = (unsigned short*)carve((size_t)L * Dm * Dm * 2);
  unsigned short* W1T = (unsigned short*)carve((size_t)L * Dm * FF * 2);
  unsigned short* W2T = (unsigned short*)carve((size_t)L * FF * Dm * 2);
  unsigned short* RKB = (unsigned short*)carve((size_t)L * NR * DK * 2);
  float*          Hb  = (float*)carve((size_t)MD * 4);
  unsigned short* XNB = (unsigned short*)carve((size_t)MD * 2);
  unsigned short* ZNB = (unsigned short*)carve((size_t)MD * 2);
  unsigned short* QB  = (unsigned short*)carve((size_t)MD * 2);
  unsigned short* KB  = (unsigned short*)carve((size_t)MD * 2);
  unsigned short* VTB = (unsigned short*)carve((size_t)MD * 2);   // [b][h*DK+d][i]
  float*          LOG = (float*)carve((size_t)B * H * S * S * 4);
  unsigned short* ATB = (unsigned short*)carve((size_t)B * H * S * S * 2);
  float*          QR  = (float*)carve((size_t)B * H * S * NR * 4);
  float*          AW  = (float*)carve((size_t)B * H * S * NR * 4);
  float*          CTX = (float*)carve((size_t)MD * 4);
  unsigned short* CTB = (unsigned short*)carve((size_t)MD * 2);
  unsigned short* MDB = (unsigned short*)carve((size_t)B * S * FF * 2);

  auto tcvt = [&](const float* src, unsigned short* dst, int K, int N) {
    dim3 g(N / 32, K / 32, L);
    transpose_cvt_kernel<<<g, 256, 0, stream>>>(src, dst, K, N);
  };
  tcvt((const float*)d_in[2],  WQT, Dm, Dm);
  tcvt((const float*)d_in[4],  WKT, Dm, Dm);
  tcvt((const float*)d_in[6],  WVT, Dm, Dm);
  tcvt((const float*)d_in[8],  WOT, Dm, Dm);
  tcvt((const float*)d_in[14], W1T, Dm, FF);
  tcvt((const float*)d_in[16], W2T, FF, Dm);
  f32_to_bf16_kernel<<<64, 256, 0, stream>>>((const float*)d_in[18], RKB, (long long)L * NR * DK);

  hipMemcpyAsync(Hb, x, (size_t)MD * sizeof(float), hipMemcpyDeviceToDevice, stream);

  auto mk = []() { GemmP p; memset(&p, 0, sizeof(p)); p.Hdiv = 1; return p; };
  auto launch_gemm = [&](const GemmP& p, int batches) {
    dim3 g((p.N + GBN - 1) / GBN, (p.M + GBM - 1) / GBM, batches);
    gemm_bf16_kernel<<<g, 256, 0, stream>>>(p);
  };

  for (int l = 0; l < L; ++l) {
    const long long lDD = (long long)l * Dm * Dm;
    // ---- sublayer 1: h = h + attn(LN1(h))
    ln_kernel<<<B * S, 256, 0, stream>>>(Hb, ln1g + (long long)l * Dm, ln1b + (long long)l * Dm,
                                         XNB, nullptr, Dm);
    {  // q / k projections: bf16 row-major outs
      GemmP p = mk();
      p.A = XNB; p.lda = Dm; p.ldb = Dm; p.M = B * S; p.N = Dm; p.K = Dm;
      p.Bm = WQT + lDD; p.bias = bq + (long long)l * Dm; p.Cb = QB; p.ldcb = Dm; launch_gemm(p, 1);
      p.Bm = WKT + lDD; p.bias = bk + (long long)l * Dm; p.Cb = KB; launch_gemm(p, 1);
    }
    {  // v projection: transposed bf16 out [b][h*DK+d][i] (batched over b)
      GemmP p = mk();
      p.A = XNB; p.lda = Dm; p.A_bs = (long long)S * Dm; p.M = S; p.N = Dm; p.K = Dm;
      p.Bm = WVT + lDD; p.ldb = Dm; p.bias = bv + (long long)l * Dm;
      p.Ct = VTB; p.ldt = S; p.Ct_bs = (long long)Dm * S;
      launch_gemm(p, B);
    }
    // logits = q @ k^T   (batched over b,h; k rows are the [N][K] operand)
    {
      GemmP p = mk();
      p.A = QB;  p.lda = Dm; p.A_bs = (long long)S * Dm; p.A_hs = DK;
      p.Bm = KB; p.ldb = Dm; p.B_bs = (long long)S * Dm; p.B_hs = DK;
      p.C = LOG; p.ldc = S;  p.C_bs = (long long)H * S * S; p.C_hs = (long long)S * S;
      p.M = S; p.N = S; p.K = DK; p.Hdiv = H;
      launch_gemm(p, B * H);
    }
    // qr = q @ rel_k^T
    {
      GemmP p = mk();
      p.A = QB;  p.lda = Dm; p.A_bs = (long long)S * Dm; p.A_hs = DK;
      p.Bm = RKB + (long long)l * NR * DK; p.ldb = DK;
      p.C = QR; p.ldc = NR; p.C_bs = (long long)H * S * NR; p.C_hs = (long long)S * NR;
      p.M = S; p.N = NR; p.K = DK; p.Hdiv = H;
      launch_gemm(p, B * H);
    }
    softmax_rel_kernel<<<B * H * S, 256, 0, stream>>>(LOG, QR, rel, ATB, AW, SCALE);
    // ctx = attn @ v  (v^T is the [N][K] operand)
    {
      GemmP p = mk();
      p.A = ATB; p.lda = S; p.A_bs = (long long)H * S * S; p.A_hs = (long long)S * S;
      p.Bm = VTB; p.ldb = S; p.B_bs = (long long)Dm * S; p.B_hs = (long long)DK * S;
      p.C = CTX; p.ldc = Dm; p.C_bs = (long long)S * Dm; p.C_hs = DK;
      p.M = S; p.N = DK; p.K = S; p.Hdiv = H;
      launch_gemm(p, B * H);
    }
    add_rv_kernel<<<B * H * S / 8, 128, 0, stream>>>(CTX, CTB, AW, rel_v + (long long)l * NR * DK);
    // h += ctx @ Wo + bo
    {
      GemmP p = mk();
      p.A = CTB; p.lda = Dm; p.Bm = WOT + lDD; p.ldb = Dm;
      p.C = Hb; p.ldc = Dm; p.bias = bo + (long long)l * Dm; p.accum = 1;
      p.M = B * S; p.N = Dm; p.K = Dm;
      launch_gemm(p, 1);
    }
    // ---- sublayer 2: h = h + W2(relu(W1(LN2(h))))
    ln_kernel<<<B * S, 256, 0, stream>>>(Hb, ln2g + (long long)l * Dm, ln2b + (long long)l * Dm,
                                         ZNB, nullptr, Dm);
    {
      GemmP p = mk();
      p.A = ZNB; p.lda = Dm; p.Bm = W1T + (long long)l * Dm * FF; p.ldb = Dm;
      p.Cb = MDB; p.ldcb = FF; p.bias = b1 + (long long)l * FF; p.relu = 1;
      p.M = B * S; p.N = FF; p.K = Dm;
      launch_gemm(p, 1);
    }
    {
      GemmP p = mk();
      p.A = MDB; p.lda = FF; p.Bm = W2T + (long long)l * FF * Dm; p.ldb = FF;
      p.C = Hb; p.ldc = Dm; p.bias = b2 + (long long)l * Dm; p.accum = 1;
      p.M = B * S; p.N = Dm; p.K = FF;
      launch_gemm(p, 1);
    }
  }
  ln_kernel<<<B * S, 256, 0, stream>>>(Hb, lnfg, lnfb, nullptr, (float*)d_out, Dm);
}
